// Qwen2MoeForCausalLM_50835232916120
// MI455X (gfx1250) — matrix operand order repaired and checked by reading, not compiled, not run on hardware
//
#include <hip/hip_runtime.h>

typedef unsigned short u16;
typedef __bf16 v16bf __attribute__((ext_vector_type(16)));
typedef float v8f __attribute__((ext_vector_type(8)));

union Frag { uint4 u[2]; v16bf v; };

static constexpr int T = 1024, H = 2048, I = 1408, E = 8;

__device__ __forceinline__ u16 f32_to_bf16(float f) {
    unsigned int u = __float_as_uint(f);
    u += 0x7FFFu + ((u >> 16) & 1u);   // round-to-nearest-even
    return (u16)(u >> 16);
}

__global__ __launch_bounds__(256) void zero_out_kernel(float* out, int n) {
    for (int i = blockIdx.x * blockDim.x + threadIdx.x; i < n; i += gridDim.x * blockDim.x)
        out[i] = 0.0f;
}

__global__ __launch_bounds__(256) void cast_bf16_kernel(const float* __restrict__ in,
                                                        u16* __restrict__ out, int n) {
    for (int i = blockIdx.x * blockDim.x + threadIdx.x; i < n; i += gridDim.x * blockDim.x)
        out[i] = f32_to_bf16(in[i]);
}

// in: [R, C] f32 row-major -> out: [C, R] bf16 row-major (R,C multiples of 32)
__global__ __launch_bounds__(256) void transpose_cast_kernel(const float* __restrict__ in,
                                                             u16* __restrict__ out,
                                                             int R, int C) {
    __shared__ float tile[32][33];
    const int x = blockIdx.x * 32 + threadIdx.x;  // C index (coalesced read)
    const int y0 = blockIdx.y * 32;               // R base
    for (int j = threadIdx.y; j < 32; j += 8)
        tile[j][threadIdx.x] = in[(size_t)(y0 + j) * C + x];
    __syncthreads();
    const int x2 = y0 + threadIdx.x;              // R index (coalesced write)
    const int y2 = blockIdx.x * 32;               // C base
    for (int j = threadIdx.y; j < 32; j += 8)
        out[(size_t)(y2 + j) * R + x2] = f32_to_bf16(tile[threadIdx.x][j]);
}

// One wave32 per token: logits -> softmax -> top-2 (no renorm over top-k).
__global__ __launch_bounds__(256) void router_kernel(const float* __restrict__ X,
                                                     const float* __restrict__ gw,
                                                     int* __restrict__ topIdx,
                                                     float* __restrict__ topW) {
    const int wave = threadIdx.x >> 5, lane = threadIdx.x & 31;
    const int t = blockIdx.x * 8 + wave;
    float acc[E];
    #pragma unroll
    for (int e = 0; e < E; ++e) acc[e] = 0.0f;
    const float* xr = X + (size_t)t * H;
    for (int h = lane; h < H; h += 32) {
        const float x = xr[h];
        const float4* g = (const float4*)(gw + (size_t)h * E);
        const float4 g0 = g[0], g1 = g[1];
        acc[0] += x * g0.x; acc[1] += x * g0.y; acc[2] += x * g0.z; acc[3] += x * g0.w;
        acc[4] += x * g1.x; acc[5] += x * g1.y; acc[6] += x * g1.z; acc[7] += x * g1.w;
    }
    #pragma unroll
    for (int off = 16; off > 0; off >>= 1)
        #pragma unroll
        for (int e = 0; e < E; ++e) acc[e] += __shfl_xor(acc[e], off, 32);
    float m = acc[0];
    #pragma unroll
    for (int e = 1; e < E; ++e) m = fmaxf(m, acc[e]);
    float p[E], s = 0.0f;
    #pragma unroll
    for (int e = 0; e < E; ++e) { p[e] = __expf(acc[e] - m); s += p[e]; }
    const float inv = 1.0f / s;
    int i1 = 0;
    #pragma unroll
    for (int e = 1; e < E; ++e) if (p[e] > p[i1]) i1 = e;
    int i2 = (i1 == 0) ? 1 : 0;
    #pragma unroll
    for (int e = 0; e < E; ++e) if (e != i1 && p[e] > p[i2]) i2 = e;
    if (lane == 0) {
        topIdx[t * 2 + 0] = i1;  topW[t * 2 + 0] = p[i1] * inv;
        topIdx[t * 2 + 1] = i2;  topW[t * 2 + 1] = p[i2] * inv;
    }
}

// Single block: bucket tokens by expert into fixed-stride segments (stride T),
// pad each segment to a multiple of 128 with (token 0, weight 0) sentinels.
__global__ __launch_bounds__(256) void build_lists_kernel(const int* __restrict__ topIdx,
                                                          const float* __restrict__ topW,
                                                          int* __restrict__ rows,
                                                          float* __restrict__ rowW,
                                                          int* __restrict__ cntPad) {
    __shared__ int sOff[E];
    const int tid = threadIdx.x;
    if (tid < E) sOff[tid] = 0;
    __syncthreads();
    for (int t = tid; t < T; t += 256)
        #pragma unroll
        for (int k = 0; k < 2; ++k) {
            const int e = topIdx[t * 2 + k];
            const int pos = atomicAdd(&sOff[e], 1);
            rows[e * T + pos] = t;
            rowW[e * T + pos] = topW[t * 2 + k];
        }
    __syncthreads();
    for (int e = 0; e < E; ++e) {
        const int c  = sOff[e];
        const int cp = (c + 127) & ~127;
        for (int pp = c + tid; pp < cp; pp += 256) {
            rows[e * T + pp] = 0;
            rowW[e * T + pp] = 0.0f;
        }
        if (tid == 0) cntPad[e] = cp;
    }
}

// C = A[M,K] x B[N,K]^T via v_wmma_f32_16x16x32_bf16, M-blocks past cntPad exit.
// EPI==0: A rows gathered via rows[]; store silu(C) bf16 to outB (expert-local rows).
// EPI==1: A rows local; scatter outF[token,col] += rowW[row] * C.
template <int EPI>
__global__ __launch_bounds__(256) void gemm_wmma_kernel(
    const u16* __restrict__ A, const u16* __restrict__ B,
    int N, int Kd,
    u16* __restrict__ outB, float* __restrict__ outF,
    const int* __restrict__ rows, const float* __restrict__ rowW,
    const int* __restrict__ cntPad, int expert) {
    constexpr int BM = 128, BN = 128, BK = 32, LS = 40;  // LS keeps rows 16B-aligned
    const int bM = blockIdx.y * BM, bN = blockIdx.x * BN;
    if (bM >= cntPad[expert]) return;   // uniform per block: EXEC stays all-ones

    __shared__ u16 As[BM * LS];
    __shared__ u16 Bs[BN * LS];
    const int tid = threadIdx.x;
    const int wave = tid >> 5, lane = tid & 31;
    const int wm = (wave & 3) * 32, wn = (wave >> 2) * 64;
    const int laneN = lane & 15, laneH = lane >> 4;
    const int* rowsE = rows + expert * T;
    const float* rowWE = rowW + expert * T;

    // per-thread LDS-fill coordinates (2 x 16B per operand)
    const int r0 = tid >> 2, r1 = r0 + 64, c8 = (tid & 3) * 8;
    const u16* aSrc0;
    const u16* aSrc1;
    if (EPI == 0) {  // gather token rows from Xb
        aSrc0 = A + (size_t)rowsE[bM + r0] * Kd + c8;
        aSrc1 = A + (size_t)rowsE[bM + r1] * Kd + c8;
    } else {         // Hact is already expert-local compact
        aSrc0 = A + (size_t)(bM + r0) * Kd + c8;
        aSrc1 = A + (size_t)(bM + r1) * Kd + c8;
    }
    const u16* bSrc0 = B + (size_t)(bN + r0) * Kd + c8;
    const u16* bSrc1 = B + (size_t)(bN + r1) * Kd + c8;

    v8f c[2][4] = {};

    for (int k0 = 0; k0 < Kd; k0 += BK) {
        __syncthreads();
        *(uint4*)&As[r0 * LS + c8] = *(const uint4*)(aSrc0 + k0);
        *(uint4*)&As[r1 * LS + c8] = *(const uint4*)(aSrc1 + k0);
        *(uint4*)&Bs[r0 * LS + c8] = *(const uint4*)(bSrc0 + k0);
        *(uint4*)&Bs[r1 * LS + c8] = *(const uint4*)(bSrc1 + k0);
        __syncthreads();

        Frag a[2], b[4];
        #pragma unroll
        for (int i = 0; i < 2; ++i) {
            const u16* p = &As[(wm + i * 16 + laneN) * LS];
            a[i].u[0] = *(const uint4*)&p[16 * laneH];       // K = 16h .. 16h+7
            a[i].u[1] = *(const uint4*)&p[16 * laneH + 8];   // K = 16h+8 .. 16h+15
        }
        #pragma unroll
        for (int j = 0; j < 4; ++j) {
            const u16* p = &Bs[(wn + j * 16 + laneN) * LS + 16 * laneH];  // K = 16h..+15
            b[j].u[0] = *(const uint4*)&p[0];
            b[j].u[1] = *(const uint4*)&p[8];
        }
        #pragma unroll
        for (int i = 0; i < 2; ++i)
            #pragma unroll
            for (int j = 0; j < 4; ++j)
                c[i][j] = __builtin_amdgcn_wmma_f32_16x16x32_bf16(
                    false, a[i].v, false, b[j].v, (short)0, c[i][j], false, false);
    }

    #pragma unroll
    for (int i = 0; i < 2; ++i)
        #pragma unroll
        for (int j = 0; j < 4; ++j) {
            const int col = bN + wn + j * 16 + laneN;
            const int rowBase = bM + wm + i * 16 + 8 * laneH;  // VGPR r -> local row rowBase+r
            #pragma unroll
            for (int r = 0; r < 8; ++r) {
                const int row = rowBase + r;
                const float v = c[i][j][r];
                if (EPI == 0) {
                    const float s = v / (1.0f + __expf(-v));  // silu
                    outB[(size_t)row * N + col] = f32_to_bf16(s);
                } else {
                    const int tok = rowsE[row];
                    const float w = rowWE[row];
                    outF[(size_t)tok * N + col] += w * v;     // serialized per-expert launches
                }
            }
        }
}

extern "C" void kernel_launch(void* const* d_in, const int* in_sizes, int n_in,
                              void* d_out, int out_size, void* d_ws, size_t ws_size,
                              hipStream_t stream) {
    const float* X  = (const float*)d_in[0];   // [T,H]
    const float* gw = (const float*)d_in[1];   // [H,E]
    const float* w1 = (const float*)d_in[2];   // [E,H,I]
    const float* w2 = (const float*)d_in[3];   // [E,I,H]
    float* out = (float*)d_out;                // [T,H]

    char* ws = (char*)d_ws;
    int*   topIdx = (int*)  (ws);                        // T*2 i32   =  8 KB
    float* topW   = (float*)(ws + 8192);                 // T*2 f32   =  8 KB
    int*   rows   = (int*)  (ws + 16384);                // E*T i32   = 32 KB
    float* rowW   = (float*)(ws + 49152);                // E*T f32   = 32 KB
    int*   cntPad = (int*)  (ws + 81920);                // E i32
    u16* Xb   = (u16*)(ws + 98304);                                  // T*H bf16 = 4 MB
    u16* w1bT = (u16*)(ws + 98304 + (size_t)T * H * 2);              // I*H bf16
    u16* w2bT = (u16*)(ws + 98304 + (size_t)T * H * 2 + (size_t)I * H * 2);
    u16* Hact = (u16*)(ws + 98304 + (size_t)T * H * 2 + 2 * (size_t)I * H * 2);  // T*I bf16

    zero_out_kernel<<<512, 256, 0, stream>>>(out, T * H);
    router_kernel<<<T / 8, 256, 0, stream>>>(X, gw, topIdx, topW);
    build_lists_kernel<<<1, 256, 0, stream>>>(topIdx, topW, rows, rowW, cntPad);
    cast_bf16_kernel<<<1024, 256, 0, stream>>>(X, Xb, T * H);

    for (int e = 0; e < E; ++e) {
        // w1[e]: [H,I] -> [I,H] bf16 ; w2[e]: [I,H] -> [H,I] bf16
        transpose_cast_kernel<<<dim3(I / 32, H / 32), dim3(32, 8), 0, stream>>>(
            w1 + (size_t)e * H * I, w1bT, H, I);
        transpose_cast_kernel<<<dim3(H / 32, I / 32), dim3(32, 8), 0, stream>>>(
            w2 + (size_t)e * I * H, w2bT, I, H);
        // Hact[local] = silu(gather(Xb) @ w1[e])    M<=T (routed), N=I, K=H
        gemm_wmma_kernel<0><<<dim3(I / 128, T / 128), 256, 0, stream>>>(
            Xb, w1bT, I, H, Hact, nullptr, rows, rowW, cntPad, e);
        // out[token] += rowW * (Hact @ w2[e])       M<=T (routed), N=H, K=I
        gemm_wmma_kernel<1><<<dim3(H / 128, T / 128), 256, 0, stream>>>(
            Hact, w2bT, H, I, nullptr, out, rows, rowW, cntPad, e);
    }
}
